// SQVarianceAdaptor_33157147525600
// MI455X (gfx1250) — compile-verified, matrix operand order checked by reading, not run on hardware
//
#include <hip/hip_runtime.h>
#include <hip/hip_bf16.h>

typedef __attribute__((ext_vector_type(16))) _Float16 v16h;
typedef __attribute__((ext_vector_type(2)))  _Float16 h2;
typedef __attribute__((ext_vector_type(8)))  float    v8f;
typedef unsigned int u32x4 __attribute__((ext_vector_type(4)));
typedef int          i32x8 __attribute__((ext_vector_type(8)));
typedef int          i32x4 __attribute__((ext_vector_type(4)));

#if defined(__gfx1250__) && __has_builtin(__builtin_amdgcn_tensor_load_to_lds)
#define HAVE_TDM 1
#else
#define HAVE_TDM 0
#endif

// ---------------- problem constants ----------------
constexpr int D_MODEL = 384;
constexpr int N_BINS  = 256;
constexpr int N_CODES = 128;
constexpr int FILT    = 256;
constexpr int NB      = 16;     // batch
constexpr int T_SRC   = 512;
constexpr int T_MEL   = 2048;

// output layout (float elements), in reference tuple order
constexpr size_t OUT_OFF     = 0;
constexpr size_t LOGDUR_OFF  = OUT_OFF     + (size_t)NB * T_MEL * D_MODEL;
constexpr size_t PITCHP_OFF  = LOGDUR_OFF  + (size_t)NB * T_SRC;
constexpr size_t ENERGYP_OFF = PITCHP_OFF  + (size_t)NB * T_MEL;
constexpr size_t MELLEN_OFF  = ENERGYP_OFF + (size_t)NB * T_MEL;
constexpr size_t MELMASK_OFF = MELLEN_OFF  + (size_t)NB * T_MEL * D_MODEL;
constexpr size_t XREG_OFF    = MELMASK_OFF + (size_t)NB * T_MEL;

// workspace layout (bytes)
constexpr size_t WS_SUME2 = 0;            // 128 f32
constexpr size_t WS_CUM   = 4096;         // 16*512 i32
constexpr size_t WS_ET    = 65536;        // 49152 f16 (swizzled codebook B-frags)
constexpr size_t WS_WTA   = 262144;       // 294912 f16 (swizzled conv1 weights)
constexpr size_t WS_WTB   = 917504;       // 196608 f16 (swizzled conv2 weights)
constexpr size_t WS_ZQ    = 2097152;      // 16*512*384 f16
constexpr size_t WS_H1    = 8388608;      // 16*2048*256 f16
constexpr size_t WS_XR16  = 25165824;     // 16*2048*384 f16

// K offset within a 32-wide chunk for register-pair rp and lane-half h
__device__ __forceinline__ int kpair_base(int rp, int h) {
    return ((rp & 3) * 2) + ((rp & 4) ? 16 : 0) + h * 8;
}

// ---------------- small prep kernels ----------------
__global__ void sume2_kernel(const float* __restrict__ E, float* __restrict__ out) {
    int m = threadIdx.x;
    if (m < N_CODES) {
        float s = 0.f;
        for (int c = 0; c < D_MODEL; ++c) { float v = E[m * D_MODEL + c]; s += v * v; }
        out[m] = s;
    }
}

// swizzle codebook into WMMA B fragments: frag[(cb*2+h)*128 + code][16 halves]
__global__ void prep_cb_kernel(const float* __restrict__ E, _Float16* __restrict__ wt) {
    int i = blockIdx.x * blockDim.x + threadIdx.x;
    if (i >= D_MODEL * N_CODES) return;
    int j  = i & 1;
    int rp = (i >> 1) & 7;
    int n  = (i >> 4) & 127;
    int h  = (i >> 11) & 1;
    int cb = i >> 12;
    int kg = cb * 32 + kpair_base(rp, h) + j;     // channel index
    wt[i] = (_Float16)E[n * D_MODEL + kg];
}

// swizzle conv weights (F=256, D_IN, 3) into B fragments over K = k*D_IN + c
template <int D_IN>
__global__ void prep_conv_kernel(const float* __restrict__ w, _Float16* __restrict__ wt) {
    int i = blockIdx.x * blockDim.x + threadIdx.x;
    if (i >= 3 * D_IN * FILT) return;
    int j  = i & 1;
    int rp = (i >> 1) & 7;
    int f  = (i >> 4) & 255;
    int h  = (i >> 12) & 1;
    int cb = i >> 13;
    int kg = cb * 32 + kpair_base(rp, h) + j;
    int c  = kg % D_IN;
    int k  = kg / D_IN;
    wt[i] = (_Float16)w[(f * D_IN + c) * 3 + k];
}

// ---------------- VQ: distances via WMMA + argmin + gather ----------------
__global__ void __launch_bounds__(256) vq_kernel(const float* __restrict__ x,
                                                 const float* __restrict__ codebook,
                                                 const float* __restrict__ sumE2,
                                                 const _Float16* __restrict__ Et,
                                                 _Float16* __restrict__ zq) {
    __shared__ _Float16 sA[16 * D_MODEL];
    __shared__ float    sS[16 * N_CODES];
    __shared__ int      sIdx[16];

    const int tid  = threadIdx.x;
    const size_t row0 = (size_t)blockIdx.x * 16;   // flattened (b,t) rows

    for (int e = tid; e < 16 * D_MODEL; e += 256) {
        int m = e / D_MODEL, c = e % D_MODEL;
        sA[e] = (_Float16)x[(row0 + m) * D_MODEL + c];
    }
    __syncthreads();

    const int wv = tid >> 5, l = tid & 31, n = l & 15, h = l >> 4;
    const int code = wv * 16 + n;

    v8f acc = {};
    for (int cb = 0; cb < D_MODEL / 32; ++cb) {
        v16h a;
#pragma unroll
        for (int rp = 0; rp < 8; ++rp) {
            int kb = cb * 32 + kpair_base(rp, h);
            h2 p = *reinterpret_cast<const h2*>(&sA[n * D_MODEL + kb]);
            a[2 * rp] = p.x; a[2 * rp + 1] = p.y;
        }
        v16h b = *reinterpret_cast<const v16h*>(Et + ((((size_t)cb * 2 + h) * N_CODES + code) << 4));
        acc = __builtin_amdgcn_wmma_f32_16x16x32_f16(false, a, false, b, (short)0, acc, false, false);
    }

    float e2 = sumE2[code];
#pragma unroll
    for (int r = 0; r < 8; ++r) {
        int m = r + 8 * h;
        sS[m * N_CODES + code] = e2 - 2.f * acc[r];
    }
    __syncthreads();

    if (tid < 16) {
        const float* row = &sS[tid * N_CODES];
        float best = row[0]; int bi = 0;
        for (int c = 1; c < N_CODES; ++c) { float v = row[c]; if (v < best) { best = v; bi = c; } }
        sIdx[tid] = bi;
    }
    __syncthreads();

    for (int e = tid; e < 16 * D_MODEL; e += 256) {
        int m = e / D_MODEL, c = e % D_MODEL;
        zq[(row0 + m) * D_MODEL + c] = (_Float16)codebook[sIdx[m] * D_MODEL + c];
    }
}

// ---------------- length regulation ----------------
__global__ void scan_kernel(const int* __restrict__ dur, int* __restrict__ cum) {
    __shared__ int s[T_SRC];
    int b = blockIdx.x, tid = threadIdx.x;
    s[tid] = dur[b * T_SRC + tid];
    __syncthreads();
    for (int off = 1; off < T_SRC; off <<= 1) {
        int v = (tid >= off) ? s[tid - off] : 0;
        __syncthreads();
        s[tid] += v;
        __syncthreads();
    }
    cum[b * T_SRC + tid] = s[tid];
}

__global__ void gather_kernel(const float* __restrict__ x, const int* __restrict__ cum,
                              float* __restrict__ xreg, float* __restrict__ mellen,
                              float* __restrict__ melmask, _Float16* __restrict__ xr16) {
    int pos = blockIdx.x, b = blockIdx.y, tid = threadIdx.x;
    const int* c = cum + b * T_SRC;
    int lo = 0, hi = T_SRC;
    while (lo < hi) { int mid = (lo + hi) >> 1; if (c[mid] <= pos) lo = mid + 1; else hi = mid; }
    int idx = lo < T_SRC - 1 ? lo : T_SRC - 1;
    bool valid = pos < c[T_SRC - 1];
    const float* src = x + ((size_t)b * T_SRC + idx) * D_MODEL;
    size_t dst = ((size_t)b * T_MEL + pos) * D_MODEL;
    for (int ch = tid; ch < D_MODEL; ch += 128) {
        float v = valid ? src[ch] : 0.f;
        xreg[dst + ch]   = v;
        mellen[dst + ch] = v;   // reference: mel_len = x_reg
        xr16[dst + ch]   = (_Float16)v;
    }
    if (tid == 0) melmask[(size_t)b * T_MEL + pos] = 1.0f;
}

// ---------------- variance predictor conv (+LN, optional fused linear) ----------------
// Raw halo rows t0-1 .. t0+16 staged in LDS (TDM when available); A[m][k*D_IN+c] = sR[(m+k)*D_IN+c].
template <int D_IN, bool FUSE_LIN>
__global__ void __launch_bounds__(512) vp_conv_kernel(
    const _Float16* __restrict__ in, int T,
    const _Float16* __restrict__ wt,
    const float* __restrict__ bias,
    const float* __restrict__ lnw, const float* __restrict__ lnb,
    const float* __restrict__ linw, const float* __restrict__ linb,
    _Float16* __restrict__ hout, float* __restrict__ sout) {
    constexpr int KW = 3 * D_IN;
    constexpr int NK = KW / 32;
    __shared__ _Float16 sR[18 * D_IN];
    __shared__ float    sH[16 * FILT];
    __shared__ float    sStat[32];

    const int t0  = blockIdx.x * 16;
    const int b   = blockIdx.y;
    const int tid = threadIdx.x;

#if HAVE_TDM
    {
        const int r0 = (t0 - 1 < 0) ? 0 : t0 - 1;
        const int r1 = (t0 + 17 > T) ? T : t0 + 17;
        const int nrows = r1 - r0;
        if (tid < 32) {
            // Tensor DMA descriptor (D#): 2-D tile of f16, nrows x D_IN, global -> LDS
            unsigned long long ga =
                (unsigned long long)(uintptr_t)(in + ((size_t)b * T + r0) * D_IN);
            unsigned lds = (unsigned)(uintptr_t)(&sR[0]) +
                           (unsigned)((r0 - (t0 - 1)) * D_IN * 2);
            u32x4 g0;
            g0[0] = 1u;                                   // count=1 valid descriptor
            g0[1] = lds;                                  // lds_addr
            g0[2] = (unsigned)(ga & 0xffffffffu);         // global_addr[31:0]
            g0[3] = (unsigned)((ga >> 32) & 0x01ffffffu)  // global_addr[56:32]
                    | (2u << 30);                         // type=2 (image)
            i32x8 g1;
            g1[0] = (int)(1u << 16);                                      // data_size=2B
            g1[1] = (int)(((unsigned)D_IN & 0xffffu) << 16);              // tensor_dim0 lo16
            g1[2] = (int)((((unsigned)D_IN) >> 16) |
                          (((unsigned)T & 0xffffu) << 16));               // dim0 hi | dim1 lo
            g1[3] = (int)((((unsigned)T) >> 16) |
                          (((unsigned)D_IN) << 16));                      // dim1 hi | tile_dim0
            g1[4] = (int)(unsigned)nrows;                                 // tile_dim1 (tile_dim2=0)
            g1[5] = (int)(unsigned)D_IN;                                  // tensor_dim0_stride
            g1[6] = 0;
            g1[7] = 0;
            i32x4 z4 = {0, 0, 0, 0};
            i32x8 z8 = {0, 0, 0, 0, 0, 0, 0, 0};
            __builtin_amdgcn_tensor_load_to_lds(g0, g1, z4, z4, z8, 0);
        }
        // zero-fill conv halo rows outside [0,T)
        if (t0 == 0)
            for (int e = tid; e < D_IN; e += 512) sR[e] = (_Float16)0.f;
        if (t0 + 16 == T)
            for (int e = tid; e < D_IN; e += 512) sR[17 * D_IN + e] = (_Float16)0.f;
        if (tid < 32) __builtin_amdgcn_s_wait_tensorcnt(0);
        __syncthreads();
    }
#else
    for (int e = tid; e < 18 * D_IN; e += 512) {
        int r = e / D_IN, c = e % D_IN;
        int t = t0 - 1 + r;
        _Float16 v = (_Float16)0.f;
        if (t >= 0 && t < T) v = in[((size_t)b * T + t) * D_IN + c];
        sR[e] = v;
    }
    __syncthreads();
#endif

    const int wv = tid >> 5, l = tid & 31, n = l & 15, h = l >> 4;
    const int f0 = wv * 16;

    v8f acc = {};
    for (int cb = 0; cb < NK; ++cb) {
        v16h a;
#pragma unroll
        for (int rp = 0; rp < 8; ++rp) {
            int kg = cb * 32 + kpair_base(rp, h);   // constexpr after full unroll
            int r  = kg / D_IN;                     // conv tap (0..2)
            int c  = kg % D_IN;                     // channel
            h2 p = *reinterpret_cast<const h2*>(&sR[(n + r) * D_IN + c]);
            a[2 * rp] = p.x; a[2 * rp + 1] = p.y;
        }
        v16h bfrag = *reinterpret_cast<const v16h*>(wt + ((((size_t)cb * 2 + h) * FILT + f0 + n) << 4));
        acc = __builtin_amdgcn_wmma_f32_16x16x32_f16(false, a, false, bfrag, (short)0, acc, false, false);
    }

    // bias + relu into tile
    float bv = bias[f0 + n];
#pragma unroll
    for (int r = 0; r < 8; ++r) {
        int m = r + 8 * h;
        float v = acc[r] + bv;
        sH[m * FILT + f0 + n] = v > 0.f ? v : 0.f;
    }
    __syncthreads();

    // per-timestep LayerNorm stats: wave wv handles t = wv
    {
        int t = wv;
        float s = 0.f, s2 = 0.f;
        for (int f = l; f < FILT; f += 32) { float v = sH[t * FILT + f]; s += v; s2 += v * v; }
        for (int off = 16; off > 0; off >>= 1) {
            s  += __shfl_xor(s,  off, 32);
            s2 += __shfl_xor(s2, off, 32);
        }
        if (l == 0) {
            float mu  = s * (1.f / FILT);
            float var = s2 * (1.f / FILT) - mu * mu;
            sStat[2 * t]     = mu;
            sStat[2 * t + 1] = rsqrtf(var + 1e-5f);
        }
    }
    __syncthreads();

    if (FUSE_LIN) {
        int t = wv;
        float mu = sStat[2 * t], is = sStat[2 * t + 1];
        float s = 0.f;
        for (int f = l; f < FILT; f += 32) {
            float hn = (sH[t * FILT + f] - mu) * is * lnw[f] + lnb[f];
            s += hn * linw[f];
        }
        for (int off = 16; off > 0; off >>= 1) s += __shfl_xor(s, off, 32);
        if (l == 0) sout[(size_t)b * T + t0 + t] = s + linb[0];
    } else {
        for (int e = tid; e < 16 * FILT; e += 512) {
            int m = e >> 8, f = e & 255;
            float hn = (sH[e] - sStat[2 * m]) * sStat[2 * m + 1] * lnw[f] + lnb[f];
            hout[((size_t)b * T + t0 + m) * FILT + f] = (_Float16)hn;
        }
    }
}

// ---------------- pitch/energy embedding + final add ----------------
__global__ void final_kernel(const float* __restrict__ xreg,
                             const float* __restrict__ pt, const float* __restrict__ et,
                             const float* __restrict__ pbins, const float* __restrict__ ebins,
                             const float* __restrict__ pemb, const float* __restrict__ eemb,
                             float* __restrict__ out) {
    int pos = blockIdx.x, b = blockIdx.y, tid = threadIdx.x;
    float pv = pt[(size_t)b * T_MEL + pos];
    float ev = et[(size_t)b * T_MEL + pos];
    int lo = 0, hi = N_BINS - 1;
    while (lo < hi) { int mid = (lo + hi) >> 1; if (pbins[mid] < pv) lo = mid + 1; else hi = mid; }
    int pidx = lo;
    lo = 0; hi = N_BINS - 1;
    while (lo < hi) { int mid = (lo + hi) >> 1; if (ebins[mid] < ev) lo = mid + 1; else hi = mid; }
    int eidx = lo;
    size_t base = ((size_t)b * T_MEL + pos) * D_MODEL;
    for (int ch = tid; ch < D_MODEL; ch += 128)
        out[base + ch] = xreg[base + ch] + pemb[pidx * D_MODEL + ch] + eemb[eidx * D_MODEL + ch];
}

// ---------------- host orchestration ----------------
extern "C" void kernel_launch(void* const* d_in, const int* in_sizes, int n_in,
                              void* d_out, int out_size, void* d_ws, size_t ws_size,
                              hipStream_t stream) {
    (void)in_sizes; (void)n_in; (void)out_size; (void)ws_size;

    const float* x        = (const float*)d_in[0];
    const int*   dur_tgt  = (const int*)d_in[3];
    const float* pitch_t  = (const float*)d_in[4];
    const float* energy_t = (const float*)d_in[5];
    const float* codebook = (const float*)d_in[6];
    // param dicts flattened in insertion order
    const float* d_c1w = (const float*)d_in[8],  *d_c1b = (const float*)d_in[9];
    const float* d_l1w = (const float*)d_in[10], *d_l1b = (const float*)d_in[11];
    const float* d_c2w = (const float*)d_in[12], *d_c2b = (const float*)d_in[13];
    const float* d_l2w = (const float*)d_in[14], *d_l2b = (const float*)d_in[15];
    const float* d_lw  = (const float*)d_in[16], *d_lb  = (const float*)d_in[17];
    const float* p_c1w = (const float*)d_in[18], *p_c1b = (const float*)d_in[19];
    const float* p_l1w = (const float*)d_in[20], *p_l1b = (const float*)d_in[21];
    const float* p_c2w = (const float*)d_in[22], *p_c2b = (const float*)d_in[23];
    const float* p_l2w = (const float*)d_in[24], *p_l2b = (const float*)d_in[25];
    const float* p_lw  = (const float*)d_in[26], *p_lb  = (const float*)d_in[27];
    const float* e_c1w = (const float*)d_in[28], *e_c1b = (const float*)d_in[29];
    const float* e_l1w = (const float*)d_in[30], *e_l1b = (const float*)d_in[31];
    const float* e_c2w = (const float*)d_in[32], *e_c2b = (const float*)d_in[33];
    const float* e_l2w = (const float*)d_in[34], *e_l2b = (const float*)d_in[35];
    const float* e_lw  = (const float*)d_in[36], *e_lb  = (const float*)d_in[37];
    const float* pitch_emb  = (const float*)d_in[38];
    const float* energy_emb = (const float*)d_in[39];
    const float* pitch_bins = (const float*)d_in[40];
    const float* energy_bins= (const float*)d_in[41];

    float* out = (float*)d_out;
    char*  ws  = (char*)d_ws;
    float*     sumE2 = (float*)(ws + WS_SUME2);
    int*       cum   = (int*)(ws + WS_CUM);
    _Float16*  Et    = (_Float16*)(ws + WS_ET);
    _Float16*  wtA   = (_Float16*)(ws + WS_WTA);
    _Float16*  wtB   = (_Float16*)(ws + WS_WTB);
    _Float16*  zq    = (_Float16*)(ws + WS_ZQ);
    _Float16*  h1    = (_Float16*)(ws + WS_H1);
    _Float16*  xr16  = (_Float16*)(ws + WS_XR16);

    // --- VQ + gather z_q ---
    sume2_kernel<<<1, 128, 0, stream>>>(codebook, sumE2);
    prep_cb_kernel<<<(D_MODEL * N_CODES + 255) / 256, 256, 0, stream>>>(codebook, Et);
    vq_kernel<<<(NB * T_SRC) / 16, 256, 0, stream>>>(x, codebook, sumE2, Et, zq);

    // --- length regulation ---
    scan_kernel<<<NB, T_SRC, 0, stream>>>(dur_tgt, cum);
    gather_kernel<<<dim3(T_MEL, NB), 128, 0, stream>>>(
        x, cum, out + XREG_OFF, out + MELLEN_OFF, out + MELMASK_OFF, xr16);

    const dim3 gS(T_SRC / 16, NB), gM(T_MEL / 16, NB);
    const int pc1 = 3 * D_MODEL * FILT, pc2 = 3 * FILT * FILT;

    // --- duration predictor (input z_q f16, T=512) ---
    prep_conv_kernel<D_MODEL><<<(pc1 + 255) / 256, 256, 0, stream>>>(d_c1w, wtA);
    vp_conv_kernel<D_MODEL, false><<<gS, 512, 0, stream>>>(
        zq, T_SRC, wtA, d_c1b, d_l1w, d_l1b, nullptr, nullptr, h1, nullptr);
    prep_conv_kernel<FILT><<<(pc2 + 255) / 256, 256, 0, stream>>>(d_c2w, wtB);
    vp_conv_kernel<FILT, true><<<gS, 512, 0, stream>>>(
        h1, T_SRC, wtB, d_c2b, d_l2w, d_l2b, d_lw, d_lb, nullptr, out + LOGDUR_OFF);

    // --- pitch predictor (input x_reg f16, T=2048) ---
    prep_conv_kernel<D_MODEL><<<(pc1 + 255) / 256, 256, 0, stream>>>(p_c1w, wtA);
    vp_conv_kernel<D_MODEL, false><<<gM, 512, 0, stream>>>(
        xr16, T_MEL, wtA, p_c1b, p_l1w, p_l1b, nullptr, nullptr, h1, nullptr);
    prep_conv_kernel<FILT><<<(pc2 + 255) / 256, 256, 0, stream>>>(p_c2w, wtB);
    vp_conv_kernel<FILT, true><<<gM, 512, 0, stream>>>(
        h1, T_MEL, wtB, p_c2b, p_l2w, p_l2b, p_lw, p_lb, nullptr, out + PITCHP_OFF);

    // --- energy predictor (input x_reg f16, T=2048) ---
    prep_conv_kernel<D_MODEL><<<(pc1 + 255) / 256, 256, 0, stream>>>(e_c1w, wtA);
    vp_conv_kernel<D_MODEL, false><<<gM, 512, 0, stream>>>(
        xr16, T_MEL, wtA, e_c1b, e_l1w, e_l1b, nullptr, nullptr, h1, nullptr);
    prep_conv_kernel<FILT><<<(pc2 + 255) / 256, 256, 0, stream>>>(e_c2w, wtB);
    vp_conv_kernel<FILT, true><<<gM, 512, 0, stream>>>(
        h1, T_MEL, wtB, e_c2b, e_l2w, e_l2b, e_lw, e_lb, nullptr, out + ENERGYP_OFF);

    // --- embeddings + final add ---
    final_kernel<<<dim3(T_MEL, NB), 128, 0, stream>>>(
        out + XREG_OFF, pitch_t, energy_t, pitch_bins, energy_bins,
        pitch_emb, energy_emb, out + OUT_OFF);
}
